// MultiHeadAttention_14379550507717
// MI455X (gfx1250) — matrix operand order repaired and checked by reading, not compiled, not run on hardware
//
#include <hip/hip_runtime.h>
#include <hip/hip_bf16.h>

// ---------------------------------------------------------------------------
// MultiHeadAttention forward (B=2,S=2048,D=1024,H=16,dk=dv=64) + LayerNorm.
// All GEMMs use v_wmma_f32_16x16x32_f16, double-buffered LDS, float4 staging,
// software-pipelined global loads + global_prefetch of the k+2 tile.
// ---------------------------------------------------------------------------

#define B_      2
#define S_      2048
#define DM_     1024
#define H_      16
#define DK_     64

typedef __attribute__((ext_vector_type(16))) _Float16 v16h;
typedef __attribute__((ext_vector_type(8)))  _Float16 v8h;
typedef __attribute__((ext_vector_type(4)))  _Float16 v4h;
typedef __attribute__((ext_vector_type(8)))  float    v8f;

#define TILE_M 64
#define TILE_N 64
#define TILE_K 32
#define LDS_ST 40   // half-element row stride; 80 B keeps 16 B alignment

// Batched GEMM: C[z] = alpha * A[z] @ op(B[z])  (+ optional mask epilogue).
// z = zb*H + zh; per-operand offset = zb*div + zh*mod.
// BT=1: B stored [N x K] row-major (multiply by its transpose, i.e. Q@K^T).
// All launches guarantee M%64==0, N%64==0, K%32==0 -> no bounds checks.
template <int BT>
__global__ __launch_bounds__(128)
void gemm_wmma_f16(const float* __restrict__ A, long long divA, long long modA, int lda,
                   const float* __restrict__ B, long long divB, long long modB, int ldb,
                   float* __restrict__ C,       long long divC, long long modC, int ldc,
                   int M, int N, int K, float alpha,
                   const unsigned char* __restrict__ mask, long long maskDiv, int H)
{
    __shared__ __align__(16) _Float16 As[2][TILE_M][LDS_ST];
    __shared__ __align__(16) _Float16 Bs[2][TILE_N][LDS_ST];   // stored [n][k]

    const int z  = blockIdx.z;
    const int zb = z / H, zh = z % H;
    A += (long long)zb * divA + (long long)zh * modA;
    B += (long long)zb * divB + (long long)zh * modB;
    C += (long long)zb * divC + (long long)zh * modC;
    const unsigned char* maskB = mask ? (mask + (long long)zb * maskDiv) : nullptr;

    const int mBase = blockIdx.y * TILE_M;
    const int nBase = blockIdx.x * TILE_N;
    const int t    = threadIdx.x;
    const int wave = t >> 5;
    const int lane = t & 31;
    const int wr = (wave >> 1) * 32;   // wave M offset inside block tile
    const int wc = (wave & 1)  * 32;   // wave N offset inside block tile

    // ---- hoisted per-thread staging addresses ----
    // A tile: 64 rows x 32 k-floats = 512 float4; thread handles rows t/8 + 16i,
    // qword column t%8 (16B-aligned since lda % 4 == 0 and k0 % 32 == 0).
    const float* aBase = A + (long long)(mBase + (t >> 3)) * lda + (t & 7) * 4;
    // B tile:
    //  BT=1: rows are n, contiguous in k -> same pattern as A.
    //  BT=0: 32 k-rows x 64 n-floats = 512 float4; thread handles k = t/16 + 8i,
    //        qword column t%16; stored transposed into LDS.
    const float* bBase = BT ? (B + (long long)(nBase + (t >> 3)) * ldb + (t & 7) * 4)
                            : (B + (long long)(t >> 4) * ldb + nBase + (t & 15) * 4);

    float4 aR[4], bR[4];

    auto loadTiles = [&](int k0) {
        #pragma unroll
        for (int i = 0; i < 4; ++i)
            aR[i] = *(const float4*)(aBase + (long long)(16 * i) * lda + k0);
        #pragma unroll
        for (int i = 0; i < 4; ++i) {
            if (BT) bR[i] = *(const float4*)(bBase + (long long)(16 * i) * ldb + k0);
            else    bR[i] = *(const float4*)(bBase + (long long)(k0 + 8 * i) * ldb);
        }
    };

    auto prefetchTiles = [&](int k0) {      // global_prefetch_b8 of the k+2 tile
        #pragma unroll
        for (int i = 0; i < 4; ++i)
            __builtin_prefetch(aBase + (long long)(16 * i) * lda + k0, 0, 3);
        #pragma unroll
        for (int i = 0; i < 4; ++i) {
            if (BT) __builtin_prefetch(bBase + (long long)(16 * i) * ldb + k0, 0, 3);
            else    __builtin_prefetch(bBase + (long long)(k0 + 8 * i) * ldb, 0, 3);
        }
    };

    auto storeTiles = [&](int buf) {
        #pragma unroll
        for (int i = 0; i < 4; ++i) {
            v4h h; h[0] = (_Float16)aR[i].x; h[1] = (_Float16)aR[i].y;
                   h[2] = (_Float16)aR[i].z; h[3] = (_Float16)aR[i].w;
            *(v4h*)&As[buf][(t >> 3) + 16 * i][(t & 7) * 4] = h;
        }
        #pragma unroll
        for (int i = 0; i < 4; ++i) {
            if (BT) {
                v4h h; h[0] = (_Float16)bR[i].x; h[1] = (_Float16)bR[i].y;
                       h[2] = (_Float16)bR[i].z; h[3] = (_Float16)bR[i].w;
                *(v4h*)&Bs[buf][(t >> 3) + 16 * i][(t & 7) * 4] = h;
            } else {
                const int k = (t >> 4) + 8 * i, n4 = (t & 15) * 4;
                Bs[buf][n4 + 0][k] = (_Float16)bR[i].x;
                Bs[buf][n4 + 1][k] = (_Float16)bR[i].y;
                Bs[buf][n4 + 2][k] = (_Float16)bR[i].z;
                Bs[buf][n4 + 3][k] = (_Float16)bR[i].w;
            }
        }
    };

    // ---- fragment addressing per ISA wave32 16-bit layouts ----
    // A 16x32: lane<16 -> row M=l, K {0..7,16..23}; lane>=16 -> K {8..15,24..31}
    // B 32x16: lane<16 -> col N=l, K {0..7,16..23}; lane>=16 -> K {8..15,24..31}
    const int kOffA = (lane >> 4) * 8;
    const int rowA  = wr + (lane & 15);
    const int kOffB = (lane >> 4) * 8;
    const int colB  = wc + (lane & 15);

    v8f acc[2][2] = {};
    const int nk = K / TILE_K;

    // prologue: stage tile 0
    loadTiles(0);
    storeTiles(0);
    __syncthreads();

    for (int kt = 0; kt < nk; ++kt) {
        const int buf = kt & 1;
        if (kt + 1 < nk) loadTiles((kt + 1) * TILE_K);       // loads in flight
        if (kt + 2 < nk) prefetchTiles((kt + 2) * TILE_K);   // warm L2/WGP$

        v16h aF[2], bF[2];
        #pragma unroll
        for (int f = 0; f < 2; ++f) {
            union { v16h v; v8h h[2]; } ua;
            ua.h[0] = *(const v8h*)&As[buf][rowA + f * 16][kOffA];
            ua.h[1] = *(const v8h*)&As[buf][rowA + f * 16][16 + kOffA];
            aF[f] = ua.v;
            union { v16h v; v8h h[2]; } ub;
            ub.h[0] = *(const v8h*)&Bs[buf][colB + f * 16][kOffB];
            ub.h[1] = *(const v8h*)&Bs[buf][colB + f * 16][16 + kOffB];
            bF[f] = ub.v;
        }
        #pragma unroll
        for (int i = 0; i < 2; ++i)
            #pragma unroll
            for (int j = 0; j < 2; ++j)
                acc[i][j] = __builtin_amdgcn_wmma_f32_16x16x32_f16(
                    false, aF[i], false, bF[j], (short)0, acc[i][j], false, false);

        if (kt + 1 < nk) storeTiles(buf ^ 1);                // fill other buffer
        __syncthreads();
    }

    // ---- epilogue: C f32 16x16 layout: VGPR v -> M=v (+8 for lanes>=16) ----
    #pragma unroll
    for (int i = 0; i < 2; ++i) {
        #pragma unroll
        for (int j = 0; j < 2; ++j) {
            const int col   = nBase + wc + j * 16 + (lane & 15);
            const int rBase = mBase + wr + i * 16 + ((lane < 16) ? 0 : 8);
            #pragma unroll
            for (int v = 0; v < 8; ++v) {
                const int r = rBase + v;
                float val = acc[i][j][v] * alpha;
                if (maskB && maskB[(long long)r * N + col]) val = -1e9f;
                C[(long long)r * ldc + col] = val;
            }
        }
    }
}

// In-place row softmax, rowLen = 2048 (512 float4), one workgroup per row.
__global__ __launch_bounds__(256)
void softmax_rows(float* __restrict__ data)
{
    __shared__ float red[256];
    float4* p = (float4*)(data + (long long)blockIdx.x * S_);
    const int t = threadIdx.x;

    float4 v0 = p[t], v1 = p[t + 256];
    float m = fmaxf(fmaxf(fmaxf(v0.x, v0.y), fmaxf(v0.z, v0.w)),
                    fmaxf(fmaxf(v1.x, v1.y), fmaxf(v1.z, v1.w)));
    red[t] = m; __syncthreads();
    for (int s = 128; s > 0; s >>= 1) { if (t < s) red[t] = fmaxf(red[t], red[t + s]); __syncthreads(); }
    m = red[0]; __syncthreads();

    v0.x = __expf(v0.x - m); v0.y = __expf(v0.y - m); v0.z = __expf(v0.z - m); v0.w = __expf(v0.w - m);
    v1.x = __expf(v1.x - m); v1.y = __expf(v1.y - m); v1.z = __expf(v1.z - m); v1.w = __expf(v1.w - m);
    float s0 = (v0.x + v0.y) + (v0.z + v0.w) + (v1.x + v1.y) + (v1.z + v1.w);
    red[t] = s0; __syncthreads();
    for (int s = 128; s > 0; s >>= 1) { if (t < s) red[t] += red[t + s]; __syncthreads(); }
    const float inv = 1.0f / red[0];

    v0.x *= inv; v0.y *= inv; v0.z *= inv; v0.w *= inv;
    v1.x *= inv; v1.y *= inv; v1.z *= inv; v1.w *= inv;
    p[t] = v0; p[t + 256] = v1;
}

// y = LayerNorm(O + inQ) * gamma + beta, d = 1024, one workgroup per row.
__global__ __launch_bounds__(256)
void residual_layernorm(const float* __restrict__ O, const float* __restrict__ inQ,
                        const float* __restrict__ gamma, const float* __restrict__ beta,
                        float* __restrict__ out)
{
    __shared__ float red[256];
    const long long row = blockIdx.x;
    const float4* o = (const float4*)(O   + row * DM_);
    const float4* q = (const float4*)(inQ + row * DM_);
    const float4* g = (const float4*)gamma;
    const float4* b = (const float4*)beta;
    float4*       y = (float4*)(out + row * DM_);
    const int t = threadIdx.x;

    float4 ov = o[t], qv = q[t];
    float4 x = make_float4(ov.x + qv.x, ov.y + qv.y, ov.z + qv.z, ov.w + qv.w);
    red[t] = (x.x + x.y) + (x.z + x.w); __syncthreads();
    for (int st = 128; st > 0; st >>= 1) { if (t < st) red[t] += red[t + st]; __syncthreads(); }
    const float mu = red[0] * (1.0f / DM_); __syncthreads();

    float4 d = make_float4(x.x - mu, x.y - mu, x.z - mu, x.w - mu);
    red[t] = (d.x * d.x + d.y * d.y) + (d.z * d.z + d.w * d.w); __syncthreads();
    for (int st = 128; st > 0; st >>= 1) { if (t < st) red[t] += red[t + st]; __syncthreads(); }
    const float rstd = rsqrtf(red[0] * (1.0f / DM_) + 1e-5f);

    float4 gv = g[t], bv = b[t];
    y[t] = make_float4(d.x * rstd * gv.x + bv.x, d.y * rstd * gv.y + bv.y,
                       d.z * rstd * gv.z + bv.z, d.w * rstd * gv.w + bv.w);
}

extern "C" void kernel_launch(void* const* d_in, const int* in_sizes, int n_in,
                              void* d_out, int out_size, void* d_ws, size_t ws_size,
                              hipStream_t stream)
{
    const float*         inQ   = (const float*)d_in[0];
    const float*         inK   = (const float*)d_in[1];
    const float*         inV   = (const float*)d_in[2];
    const unsigned char* mask  = (const unsigned char*)d_in[3];   // jnp bool -> 1 byte
    const float*         WQ    = (const float*)d_in[4];
    const float*         WK    = (const float*)d_in[5];
    const float*         WV    = (const float*)d_in[6];
    const float*         WO    = (const float*)d_in[7];
    const float*         gamma = (const float*)d_in[8];
    const float*         beta  = (const float*)d_in[9];

    const long long MS = (long long)B_ * S_;           // 4096 rows
    const long long SD = MS * DM_;                     // 4,194,304
    const long long SS = (long long)S_ * S_;           // per-(b,h) attn plane

    float* out  = (float*)d_out;
    float* attn = out + SD;                            // outputs: [out | attn]

    float* Qp  = (float*)d_ws;                         // [4096,1024] each
    float* Kp  = Qp  + SD;
    float* Vp  = Kp  + SD;
    float* Ctx = Vp  + SD;
    float* Op  = Ctx + SD;

    const dim3 blk(128);
    const dim3 gProj(DM_ / TILE_N, MS / TILE_M, 1);    // 16 x 64

    // Q/K/V projections: [4096,1024] @ [1024,1024]
    gemm_wmma_f16<0><<<gProj, blk, 0, stream>>>(inQ, 0, 0, DM_, WQ, 0, 0, DM_,
                                                Qp, 0, 0, DM_, (int)MS, DM_, DM_, 1.0f,
                                                nullptr, 0, 1);
    gemm_wmma_f16<0><<<gProj, blk, 0, stream>>>(inK, 0, 0, DM_, WK, 0, 0, DM_,
                                                Kp, 0, 0, DM_, (int)MS, DM_, DM_, 1.0f,
                                                nullptr, 0, 1);
    gemm_wmma_f16<0><<<gProj, blk, 0, stream>>>(inV, 0, 0, DM_, WV, 0, 0, DM_,
                                                Vp, 0, 0, DM_, (int)MS, DM_, DM_, 1.0f,
                                                nullptr, 0, 1);

    // scores = Q @ K^T / sqrt(dk), masked -> attn   (z = b*H + h, 32 batches)
    gemm_wmma_f16<1><<<dim3(S_ / TILE_N, S_ / TILE_M, B_ * H_), blk, 0, stream>>>(
        Qp,   (long long)S_ * DM_, DK_, DM_,
        Kp,   (long long)S_ * DM_, DK_, DM_,
        attn, (long long)H_ * SS,  SS,  S_,
        S_, S_, DK_, 0.125f, mask, SS, H_);

    // softmax over each of the B*H*S rows of attn
    softmax_rows<<<dim3((unsigned)(B_ * H_ * S_)), dim3(256), 0, stream>>>(attn);

    // context = attn @ V  -> Ctx laid out as [b, s, h*dv] (row stride 1024)
    gemm_wmma_f16<0><<<dim3(DK_ / TILE_N, S_ / TILE_M, B_ * H_), blk, 0, stream>>>(
        attn, (long long)H_ * SS,  SS,  S_,
        Vp,   (long long)S_ * DM_, DK_, DM_,
        Ctx,  (long long)S_ * DM_, DK_, DM_,
        S_, DK_, S_, 1.0f, nullptr, 0, H_);

    // out-projection: [4096,1024] @ [1024,1024]
    gemm_wmma_f16<0><<<gProj, blk, 0, stream>>>(Ctx, 0, 0, DM_, WO, 0, 0, DM_,
                                                Op, 0, 0, DM_, (int)MS, DM_, DM_, 1.0f,
                                                nullptr, 0, 1);

    // residual + LayerNorm -> out
    residual_layernorm<<<dim3((unsigned)MS), dim3(256), 0, stream>>>(Op, inQ, gamma, beta, out);
}
